// KANG_64338610094086
// MI455X (gfx1250) — matrix operand order repaired and checked by reading, not compiled, not run on hardware
//
#include <hip/hip_runtime.h>
#include <hip/hip_bf16.h>
#include <math.h>

#define N_NODES 50000
#define N_EDGES 800000
#define IN_DIM  512
#define HDIM    64
#define GRID_K  10
#define LN_EPS  1e-5f

typedef __attribute__((ext_vector_type(16))) __bf16 v16bf;
typedef __attribute__((ext_vector_type(8)))  __bf16 v8bf;
typedef __attribute__((ext_vector_type(8)))  float  v8f;
typedef __attribute__((ext_vector_type(4)))  unsigned v4u;
typedef __attribute__((ext_vector_type(8)))  int    v8i;
typedef __attribute__((ext_vector_type(4)))  int    v4i;

__device__ __forceinline__ float sigmoidf_(float x) { return 1.0f / (1.0f + __expf(-x)); }

// ---------------------------------------------------------------------------
// Kernel 1: h0 = relu(LayerNorm(x @ W0 + b0))
// 16-row tile per block; 4 waves, each wave computes a 16x16 column tile.
// WMMA bf16 (K=32 per step, 16 steps over IN_DIM=512).
// ---------------------------------------------------------------------------
__global__ __launch_bounds__(128)
void k_in_gemm(const float* __restrict__ x, const float* __restrict__ W0,
               const float* __restrict__ b0, const float* __restrict__ g,
               const float* __restrict__ bb, float* __restrict__ hout) {
    __shared__ float ct[16][68];
    __shared__ float smu[16], srs[16];
    const int tid  = threadIdx.x;
    const int wid  = tid >> 5;
    const int lane = tid & 31;
    const int m    = lane & 15;
    const int hi   = lane >> 4;
    const int n0   = wid * 16;
    const int row0 = blockIdx.x * 16;

    v8f c = {};
    for (int kb = 0; kb < IN_DIM; kb += 32) {
        v16bf a, b;
        // A fragment: lane m holds row (row0+m); element e -> k = hi*8 + (e<8 ? e : 16+e-8)
        const float* xr = x + (size_t)(row0 + m) * IN_DIM + kb + hi * 8;
        __builtin_prefetch((const void*)(xr + 32), 0, 0);   // next K tile
        float4 p0 = ((const float4*)xr)[0];
        float4 p1 = ((const float4*)xr)[1];
        float4 p2 = ((const float4*)(xr + 16))[0];
        float4 p3 = ((const float4*)(xr + 16))[1];
        a[0] = (__bf16)p0.x; a[1] = (__bf16)p0.y; a[2] = (__bf16)p0.z; a[3] = (__bf16)p0.w;
        a[4] = (__bf16)p1.x; a[5] = (__bf16)p1.y; a[6] = (__bf16)p1.z; a[7] = (__bf16)p1.w;
        a[8] = (__bf16)p2.x; a[9] = (__bf16)p2.y; a[10] = (__bf16)p2.z; a[11] = (__bf16)p2.w;
        a[12] = (__bf16)p3.x; a[13] = (__bf16)p3.y; a[14] = (__bf16)p3.z; a[15] = (__bf16)p3.w;
        // B fragment: lane holds column n0+m; element e -> k = hi*8 + (e<8 ? e : 16+e-8)
        const float* wr = W0 + (size_t)(kb + hi * 8) * HDIM + n0 + m;
        #pragma unroll
        for (int s = 0; s < 16; ++s) b[s] = (__bf16)wr[(s < 8 ? s : s + 8) * HDIM];
        c = __builtin_amdgcn_wmma_f32_16x16x32_bf16(false, a, false, b, (short)0, c,
                                                    false, false);
    }
    // C layout: VGPR r -> row (r + 8*hi), col n0 + (lane&15)
    #pragma unroll
    for (int r = 0; r < 8; ++r)
        ct[r + 8 * hi][n0 + m] = c[r] + b0[n0 + m];
    __syncthreads();

    if (tid < 16) {
        float s = 0.f;
        #pragma unroll
        for (int n = 0; n < HDIM; ++n) s += ct[tid][n];
        float mu = s * (1.0f / 64.0f);
        float v2 = 0.f;
        #pragma unroll
        for (int n = 0; n < HDIM; ++n) { float d = ct[tid][n] - mu; v2 += d * d; }
        smu[tid] = mu;
        srs[tid] = rsqrtf(v2 * (1.0f / 64.0f) + LN_EPS);
    }
    __syncthreads();

    int mr = tid >> 3, nb = (tid & 7) * 8;
    float mu = smu[mr], rs = srs[mr];
    #pragma unroll
    for (int j = 0; j < 8; ++j) {
        int n = nb + j;
        float v = (ct[mr][n] - mu) * rs * g[n] + bb[n];
        hout[(size_t)(row0 + mr) * HDIM + n] = fmaxf(v, 0.f);
    }
}

// ---------------------------------------------------------------------------
// Prep: Wa1 (128x64 f32) -> bf16 once; TDM broadcasts it to LDS in k_edge.
// ---------------------------------------------------------------------------
__global__ void k_prep(const float* __restrict__ Wa1, __bf16* __restrict__ Wa1b) {
    int i = blockIdx.x * blockDim.x + threadIdx.x;
    if (i < 2 * HDIM * HDIM) Wa1b[i] = (__bf16)Wa1[i];
}

// ---------------------------------------------------------------------------
// Kernel 2: per-edge KAN message + attention MLP (WMMA) + raw score.
// 16 edges per wave; two lanes per edge (32 dims each).
// Wa1(bf16) is DMA'd into LDS once per block via the Tensor Data Mover.
// ---------------------------------------------------------------------------
__global__ __launch_bounds__(128)
void k_edge(const float* __restrict__ h, const int* __restrict__ src,
            const int* __restrict__ dst, const int* __restrict__ etype,
            const float* __restrict__ w_base, const float* __restrict__ w_spline,
            const float* __restrict__ coeffs, const __bf16* __restrict__ Wa1b,
            const float* __restrict__ ba1, const float* __restrict__ Wa2,
            const float* __restrict__ ba2,
            float* __restrict__ msg, float* __restrict__ score) {
    // [wave][edge][2H], row stride 144 (=288B, 16B multiple) for aligned b128 loads
    __shared__ __align__(16) __bf16 comb[4][16][144];
    __shared__ __align__(32) __bf16 wlds[2 * HDIM * HDIM];   // 16KB: Wa1 in bf16

    const int tid  = threadIdx.x, wid = tid >> 5, lane = tid & 31;
    const int i    = lane & 15;           // edge-in-tile
    const int hl   = lane >> 4;           // dim half
    const int e    = blockIdx.x * 64 + wid * 16 + i;
    const int sn   = src[e], dn = dst[e], rel = etype[e];
    const int db   = hl * 32;

    // --- TDM: DMA Wa1b (2D tensor 128x64, bf16) into LDS, issued once per block
    if (tid == 0) {
        unsigned laddr = (unsigned)(size_t)(void*)wlds;          // LDS byte offset
        unsigned long long ga = (unsigned long long)(size_t)Wa1b;
        v4u g0;
        g0[0] = 1u;                                              // count=1, user mode
        g0[1] = laddr;                                           // lds_addr
        g0[2] = (unsigned)(ga & 0xFFFFFFFFu);                    // global_addr[31:0]
        g0[3] = (unsigned)((ga >> 32) & 0x01FFFFFFu) | (2u << 30); // addr[56:32] | type=2
        v8i g1;
        g1[0] = (1 << 16);            // data_size=1 (2 bytes/elem)
        g1[1] = (int)(64u << 16);     // tensor_dim0 = 64 (bits 79:48)
        g1[2] = (int)(128u << 16);    // tensor_dim1 = 128 (bits 111:80)
        g1[3] = (int)(64u << 16);     // tile_dim0 = 64 (bits 127:112)
        g1[4] = 128;                  // tile_dim1 = 128, tile_dim2 = 0
        g1[5] = 64;                   // tensor_dim0_stride = 64
        g1[6] = 0;
        g1[7] = 0;
        v4i z4 = {0, 0, 0, 0};
        v8i z8 = {0, 0, 0, 0, 0, 0, 0, 0};
        __builtin_amdgcn_tensor_load_to_lds(g0, g1, z4, z4, z8, 0);
        __builtin_amdgcn_s_wait_tensorcnt(0);
    }

    // --- gather h_src half, partial mean
    float hs[32];
    float sum = 0.f;
    const float* hp = h + (size_t)sn * HDIM + db;
    #pragma unroll
    for (int j = 0; j < 32; ++j) { hs[j] = hp[j]; sum += hs[j]; }
    sum += __shfl_xor(sum, 16, 32);
    float xm = sum * (1.0f / 64.0f);

    // --- KAN spline: softmax over truncated cubic basis
    const float dknot = 10.0f / 9.0f;
    float braw[GRID_K];
    float bmax = -1e30f;
    #pragma unroll
    for (int j = 0; j < GRID_K; ++j) {
        float knot = -5.0f + dknot * (float)j;
        float t = fminf(fmaxf((xm - knot) / dknot, 0.f), 1.f);
        float u = 1.f - t;
        float v = u * u * u * ((t < 1.f) ? 1.f : 0.f);
        braw[j] = v;
        bmax = fmaxf(bmax, v);
    }
    float bs = 0.f;
    #pragma unroll
    for (int j = 0; j < GRID_K; ++j) { braw[j] = __expf(braw[j] - bmax); bs += braw[j]; }
    float sval = 0.f;
    const float* cf = coeffs + rel * GRID_K;
    #pragma unroll
    for (int j = 0; j < GRID_K; ++j) sval += braw[j] * cf[j];
    sval *= w_spline[rel] / bs;

    float gate = sigmoidf_(xm);
    float wb   = w_base[rel];

    // --- msg (global, needed for aggregation) + comb tile [h_dst | msg] in bf16 LDS
    const float* hd = h + (size_t)dn * HDIM + db;
    float* mp = msg + (size_t)e * HDIM + db;
    #pragma unroll
    for (int j = 0; j < 32; ++j) {
        float mv = gate * wb * sigmoidf_(hs[j]) + (1.f - gate) * sval;
        mp[j] = mv;
        comb[wid][i][db + j]        = (__bf16)hd[j];
        comb[wid][i][HDIM + db + j] = (__bf16)mv;
    }
    __syncthreads();   // comb visible wave-wide; TDM data visible block-wide

    // --- hid = relu(comb @ Wa1 + ba1); score = hid @ Wa2 fused from C fragments
    float sc[8];
    #pragma unroll
    for (int rr = 0; rr < 8; ++rr) sc[rr] = 0.f;

    #pragma unroll
    for (int nt = 0; nt < 4; ++nt) {
        v8f c = {};
        #pragma unroll
        for (int kb = 0; kb < 2 * HDIM; kb += 32) {
            // A fragment: two contiguous 8-element bf16 runs -> 2x ds_load_b128
            const __bf16* cr = &comb[wid][i][kb + hl * 8];
            v8bf alo = *(const v8bf*)cr;
            v8bf ahi = *(const v8bf*)(cr + 16);
            v16bf a = __builtin_shufflevector(alo, ahi, 0, 1, 2, 3, 4, 5, 6, 7,
                                              8, 9, 10, 11, 12, 13, 14, 15);
            // B fragment from TDM-resident LDS copy of Wa1
            const __bf16* wr = wlds + (size_t)(kb + hl * 8) * HDIM + nt * 16 + i;
            v16bf b;
            #pragma unroll
            for (int s = 0; s < 16; ++s) b[s] = wr[(s < 8 ? s : s + 8) * HDIM];
            c = __builtin_amdgcn_wmma_f32_16x16x32_bf16(false, a, false, b, (short)0, c,
                                                        false, false);
        }
        int n    = nt * 16 + i;
        float a2 = Wa2[n];
        float b1 = ba1[n];
        #pragma unroll
        for (int rr = 0; rr < 8; ++rr) {
            // row (rr + 8*hl) lives across the 16 lanes of this half: butterfly over N
            float hv = fmaxf(c[rr] + b1, 0.f) * a2;
            hv += __shfl_xor(hv, 1, 32);
            hv += __shfl_xor(hv, 2, 32);
            hv += __shfl_xor(hv, 4, 32);
            hv += __shfl_xor(hv, 8, 32);
            sc[rr] += hv;
        }
    }
    if (i == 0) {
        #pragma unroll
        for (int rr = 0; rr < 8; ++rr)
            score[blockIdx.x * 64 + wid * 16 + 8 * hl + rr] = sc[rr] + ba2[0];
    }
}

// ---------------------------------------------------------------------------
// Segment softmax + aggregation helpers
// ---------------------------------------------------------------------------
__device__ __forceinline__ unsigned enc_f32(float f) {
    unsigned u = __float_as_uint(f);
    return (u & 0x80000000u) ? ~u : (u | 0x80000000u);
}
__device__ __forceinline__ float dec_f32(unsigned u) {
    return (u & 0x80000000u) ? __uint_as_float(u & 0x7FFFFFFFu) : __uint_as_float(~u);
}

__global__ void k_zero(float* __restrict__ agg, unsigned* __restrict__ nmax,
                       float* __restrict__ denom) {
    int idx = blockIdx.x * blockDim.x + threadIdx.x;
    if (idx < N_NODES * HDIM) agg[idx] = 0.f;
    if (idx < N_NODES) { nmax[idx] = 0u; denom[idx] = 0.f; }
}

__global__ void k_max(const float* __restrict__ score, const int* __restrict__ dst,
                      unsigned* __restrict__ nmax) {
    int e = blockIdx.x * blockDim.x + threadIdx.x;
    if (e >= N_EDGES) return;
    atomicMax(&nmax[dst[e]], enc_f32(score[e]));
}

__global__ void k_expsum(const float* __restrict__ score, const int* __restrict__ dst,
                         const unsigned* __restrict__ nmax, float* __restrict__ esc,
                         float* __restrict__ denom) {
    int e = blockIdx.x * blockDim.x + threadIdx.x;
    if (e >= N_EDGES) return;
    int dn = dst[e];
    float ev = __expf(score[e] - dec_f32(nmax[dn]));
    esc[e] = ev;
    atomicAdd(&denom[dn], ev);
}

__global__ void k_agg(const float* __restrict__ msg, const int* __restrict__ dst,
                      const float* __restrict__ esc, const float* __restrict__ denom,
                      float* __restrict__ agg) {
    long long gid = (long long)blockIdx.x * blockDim.x + threadIdx.x;
    if (gid >= (long long)N_EDGES * HDIM) return;
    int e  = (int)(gid >> 6);
    int dd = (int)(gid & 63);
    int dn = dst[e];
    float w = esc[e] / denom[dn];
    atomicAdd(&agg[(size_t)dn * HDIM + dd], w * msg[gid]);
}

// ---------------------------------------------------------------------------
// Kernel 4: h' = relu(LayerNorm(h + agg)); one wave per node, shuffle reduce.
// ---------------------------------------------------------------------------
__global__ __launch_bounds__(128)
void k_ln(const float* __restrict__ h, const float* __restrict__ agg,
          const float* __restrict__ g, const float* __restrict__ b,
          float* __restrict__ out) {
    int wid = threadIdx.x >> 5, lane = threadIdx.x & 31;
    int node = blockIdx.x * 4 + wid;
    if (node >= N_NODES) return;
    const float* hp = h + (size_t)node * HDIM;
    const float* ap = agg + (size_t)node * HDIM;
    float v0 = hp[2 * lane] + ap[2 * lane];
    float v1 = hp[2 * lane + 1] + ap[2 * lane + 1];
    float s = v0 + v1;
    #pragma unroll
    for (int off = 16; off >= 1; off >>= 1) s += __shfl_xor(s, off, 32);
    float mu = s * (1.0f / 64.0f);
    float d0 = v0 - mu, d1 = v1 - mu;
    float vv = d0 * d0 + d1 * d1;
    #pragma unroll
    for (int off = 16; off >= 1; off >>= 1) vv += __shfl_xor(vv, off, 32);
    float rs = rsqrtf(vv * (1.0f / 64.0f) + LN_EPS);
    out[(size_t)node * HDIM + 2 * lane]     = fmaxf(d0 * rs * g[2 * lane] + b[2 * lane], 0.f);
    out[(size_t)node * HDIM + 2 * lane + 1] = fmaxf(d1 * rs * g[2 * lane + 1] + b[2 * lane + 1], 0.f);
}

// ---------------------------------------------------------------------------
extern "C" void kernel_launch(void* const* d_in, const int* in_sizes, int n_in,
                              void* d_out, int out_size, void* d_ws, size_t ws_size,
                              hipStream_t stream) {
    const float* x       = (const float*)d_in[0];
    const int*   eidx    = (const int*)d_in[1];   // [2, E]
    const int*   etyp    = (const int*)d_in[2];
    const float* W0      = (const float*)d_in[3];
    const float* b0      = (const float*)d_in[4];
    const float* ln0g    = (const float*)d_in[5];
    const float* ln0b    = (const float*)d_in[6];
    const float* w_base  = (const float*)d_in[7];
    const float* w_spl   = (const float*)d_in[8];
    const float* coeffs  = (const float*)d_in[9];
    const float* Wa1     = (const float*)d_in[10];
    const float* ba1     = (const float*)d_in[11];
    const float* Wa2     = (const float*)d_in[12];
    const float* ba2     = (const float*)d_in[13];
    const float* lng     = (const float*)d_in[14];
    const float* lnb     = (const float*)d_in[15];
    const int* src = eidx;
    const int* dst = eidx + N_EDGES;

    char* ws = (char*)d_ws;
    size_t o = 0;
    auto carve = [&](size_t bytes) -> void* {
        void* p = ws + o;
        o = (o + bytes + 255) & ~(size_t)255;
        return p;
    };
    float*    hA    = (float*)carve((size_t)N_NODES * HDIM * 4);
    float*    hB    = (float*)carve((size_t)N_NODES * HDIM * 4);
    float*    msg   = (float*)carve((size_t)N_EDGES * HDIM * 4);
    float*    score = (float*)carve((size_t)N_EDGES * 4);
    float*    esc   = (float*)carve((size_t)N_EDGES * 4);
    unsigned* nmax  = (unsigned*)carve((size_t)N_NODES * 4);
    float*    denom = (float*)carve((size_t)N_NODES * 4);
    float*    agg   = (float*)carve((size_t)N_NODES * HDIM * 4);
    __bf16*   Wa1b  = (__bf16*)carve((size_t)2 * HDIM * HDIM * 2);

    k_prep<<<(2 * HDIM * HDIM + 255) / 256, 256, 0, stream>>>(Wa1, Wa1b);
    k_in_gemm<<<N_NODES / 16, 128, 0, stream>>>(x, W0, b0, ln0g, ln0b, hA);

    float* hcur = hA;
    for (int layer = 0; layer < 2; ++layer) {
        float* hnext = (layer == 0) ? hB : (float*)d_out;
        k_zero<<<(N_NODES * HDIM + 255) / 256, 256, 0, stream>>>(agg, nmax, denom);
        k_edge<<<N_EDGES / 64, 128, 0, stream>>>(hcur, src, dst, etyp, w_base, w_spl,
                                                 coeffs, Wa1b, ba1, Wa2, ba2, msg, score);
        k_max<<<(N_EDGES + 255) / 256, 256, 0, stream>>>(score, dst, nmax);
        k_expsum<<<(N_EDGES + 255) / 256, 256, 0, stream>>>(score, dst, nmax, esc, denom);
        long long aggw = (long long)N_EDGES * HDIM;
        k_agg<<<(unsigned)((aggw + 255) / 256), 256, 0, stream>>>(msg, dst, esc, denom, agg);
        k_ln<<<(N_NODES + 3) / 4, 128, 0, stream>>>(hcur, agg, lng + layer * HDIM,
                                                    lnb + layer * HDIM, hnext);
        hcur = hnext;
    }
}
